// MLP_45002667327743
// MI455X (gfx1250) — compile-verified, matrix operand order checked by reading, not run on hardware
//
#include <hip/hip_runtime.h>
#include <hip/hip_bf16.h>

// ---------------------------------------------------------------------------
// Fused MoE-MLP for MI455X (gfx1250, wave32).
//   h   = gelu_tanh(x @ w1)   [per expert]
//   out = h @ w2
// fp32 inputs are Dekker-split into bf16 hi/lo pairs; each K=32 step does
//   acc += lo*hi + hi*lo + hi*hi   via v_wmma_f32_16x16x32_bf16
// (~fp32 accuracy at 3 WMMA units per K=32 vs 8 units for f32 16x16x4 WMMA).
// Compute-bound: 275 GFLOP vs ~16.5us of minimum HBM traffic at 23.3 TB/s.
// ---------------------------------------------------------------------------

typedef __attribute__((ext_vector_type(16))) __bf16 v16bf;
typedef __attribute__((ext_vector_type(8)))  __bf16 v8bf;
typedef __attribute__((ext_vector_type(4)))  __bf16 v4bf;   // 8-byte packed store
typedef __attribute__((ext_vector_type(8)))  float  v8f;
typedef __attribute__((ext_vector_type(4)))  float  v4f;

#define BK  32   // K per WMMA step
#define BKP 40   // padded LDS stride in elements (80 bytes, 16B aligned)

__device__ __forceinline__ float gelu_tanh(float x) {
  const float c = 0.7978845608028654f;  // sqrt(2/pi)
  float u = c * (x + 0.044715f * x * x * x);
  return 0.5f * x * (1.0f + tanhf(u));
}

// 16 contiguous bf16 from LDS as two 16-byte loads (ds_load_b128 x2).
__device__ __forceinline__ v16bf ld_frag(const __bf16* p) {
  v8bf a = *(const v8bf*)(p);
  v8bf b = *(const v8bf*)(p + 8);
  v16bf r;
#pragma unroll
  for (int i = 0; i < 8; ++i) { r[i] = a[i]; r[i + 8] = b[i]; }
  return r;
}

// Split 4 floats into packed bf16 hi/lo quads (Dekker split, RNE).
__device__ __forceinline__ void split4(const float* f, v4bf& hi, v4bf& lo) {
#pragma unroll
  for (int j = 0; j < 4; ++j) {
    __bf16 h = (__bf16)f[j];
    hi[j] = h;
    lo[j] = (__bf16)(f[j] - (float)h);
  }
}

// C = A[MxK] * B[KxN] (+ optional fused tanh-GELU), all row-major fp32.
// Block: 256 threads = 8 waves laid out WROWS x WCOLS; each wave owns a
// (WMT*16) x (WNT*16) accumulator patch.
template <int BM, int BN, int WROWS, int WCOLS, int WMT, int WNT, bool GELU>
__global__ __launch_bounds__(256)
void mlp_gemm_bf16x3(const float* __restrict__ A, const float* __restrict__ B,
                     float* __restrict__ C, int M, int N, int K)
{
  static_assert(WROWS * WMT * 16 == BM, "M tiling mismatch");
  static_assert(WCOLS * WNT * 16 == BN, "N tiling mismatch");
  static_assert(WROWS * WCOLS == 8, "8 waves per block");

  __shared__ alignas(16) __bf16 AsHi[BM * BKP];
  __shared__ alignas(16) __bf16 AsLo[BM * BKP];
  __shared__ alignas(16) __bf16 BsHi[BN * BKP];  // stored transposed: [n][k]
  __shared__ alignas(16) __bf16 BsLo[BN * BKP];

  const int tid   = threadIdx.x;
  const int lane  = tid & 31;
  const int wid   = tid >> 5;
  const int wrow  = wid % WROWS;
  const int wcol  = wid / WROWS;
  const int lmod  = lane & 15;   // row (A) / col (B) within 16x16 tile
  const int lhalf = lane >> 4;   // K-half (0: K=0..15, 1: K=16..31)

  const int m0 = blockIdx.y * BM;
  const int n0 = blockIdx.x * BN;

  v8f acc[WMT][WNT];
#pragma unroll
  for (int mi = 0; mi < WMT; ++mi)
#pragma unroll
    for (int ni = 0; ni < WNT; ++ni)
#pragma unroll
      for (int i = 0; i < 8; ++i) acc[mi][ni][i] = 0.0f;

  for (int kt = 0; kt < K; kt += BK) {
    // ---- A tile (BM x 32): float4 per thread, fully coalesced ----
    v4f areg[BM / 32];
#pragma unroll
    for (int i = 0; i < BM / 32; ++i) {
      int idx = tid + i * 256;
      int r = idx >> 3, c4 = idx & 7;            // BM rows x 8 float4/row
      areg[i] = *(const v4f*)(A + (size_t)(m0 + r) * K + kt + c4 * 4);
    }
    // ---- B tile (32 x BN): each thread owns a 4-deep K-quad of one column.
    // Consecutive lanes -> consecutive n, so each of the 4 strided b32 loads
    // is a coalesced 128B/wave transaction; enables b64 transposed LDS stores.
    v4f breg[BN / 32];
#pragma unroll
    for (int i = 0; i < BN / 32; ++i) {
      int q  = tid + i * 256;
      int n  = q % BN;
      int k4 = (q / BN) * 4;
#pragma unroll
      for (int j = 0; j < 4; ++j)
        breg[i][j] = B[(size_t)(kt + k4 + j) * N + n0 + n];
    }
    // prefetch next K tile while we compute this one (global_prefetch_b8)
    if (kt + BK < K) {
      int r = tid >> 3, c4 = tid & 7;
      __builtin_prefetch(A + (size_t)(m0 + r) * K + (kt + BK) + c4 * 4, 0, 1);
      int nb = tid % BN, kb = (tid / BN) * 4;
      __builtin_prefetch(B + (size_t)(kt + BK + kb) * N + n0 + nb, 0, 1);
    }

    __syncthreads();  // previous iteration's LDS reads are done

    // ---- registers -> LDS, fp32 = bf16hi + bf16lo, packed b64 stores ----
#pragma unroll
    for (int i = 0; i < BM / 32; ++i) {
      int idx = tid + i * 256;
      int r = idx >> 3, c = (idx & 7) * 4;
      v4bf hi, lo;
      float f[4] = {areg[i][0], areg[i][1], areg[i][2], areg[i][3]};
      split4(f, hi, lo);
      *(v4bf*)&AsHi[r * BKP + c] = hi;
      *(v4bf*)&AsLo[r * BKP + c] = lo;
    }
#pragma unroll
    for (int i = 0; i < BN / 32; ++i) {
      int q  = tid + i * 256;
      int n  = q % BN;
      int k4 = (q / BN) * 4;
      v4bf hi, lo;
      float f[4] = {breg[i][0], breg[i][1], breg[i][2], breg[i][3]};
      split4(f, hi, lo);
      *(v4bf*)&BsHi[n * BKP + k4] = hi;   // transposed: Bs[n][k]
      *(v4bf*)&BsLo[n * BKP + k4] = lo;
    }
    __syncthreads();

    // ---- LDS -> fragments (contiguous k = 16*lhalf + s, same order in A and B) ----
    v16bf aHi[WMT], aLo[WMT], bHi[WNT], bLo[WNT];
#pragma unroll
    for (int mi = 0; mi < WMT; ++mi) {
      int row = wrow * WMT * 16 + mi * 16 + lmod;
      aHi[mi] = ld_frag(&AsHi[row * BKP + lhalf * 16]);
      aLo[mi] = ld_frag(&AsLo[row * BKP + lhalf * 16]);
    }
#pragma unroll
    for (int ni = 0; ni < WNT; ++ni) {
      int col = wcol * WNT * 16 + ni * 16 + lmod;
      bHi[ni] = ld_frag(&BsHi[col * BKP + lhalf * 16]);
      bLo[ni] = ld_frag(&BsLo[col * BKP + lhalf * 16]);
    }

    // ---- 3-term split-precision WMMA: lo*hi + hi*lo + hi*hi ----
#pragma unroll
    for (int mi = 0; mi < WMT; ++mi)
#pragma unroll
      for (int ni = 0; ni < WNT; ++ni) {
        acc[mi][ni] = __builtin_amdgcn_wmma_f32_16x16x32_bf16(
            false, aLo[mi], false, bHi[ni], (short)0, acc[mi][ni], false, false);
        acc[mi][ni] = __builtin_amdgcn_wmma_f32_16x16x32_bf16(
            false, aHi[mi], false, bLo[ni], (short)0, acc[mi][ni], false, false);
        acc[mi][ni] = __builtin_amdgcn_wmma_f32_16x16x32_bf16(
            false, aHi[mi], false, bHi[ni], (short)0, acc[mi][ni], false, false);
      }
  }

  // ---- epilogue: (GELU) + fp32 store. C/D layout: VGPR r -> M = r + 8*lhalf,
  // lane%16 -> N. ----
#pragma unroll
  for (int mi = 0; mi < WMT; ++mi)
#pragma unroll
    for (int ni = 0; ni < WNT; ++ni) {
      int mbase = m0 + wrow * WMT * 16 + mi * 16 + lhalf * 8;
      int n     = n0 + wcol * WNT * 16 + ni * 16 + lmod;
#pragma unroll
      for (int r = 0; r < 8; ++r) {
        float v = acc[mi][ni][r];
        if (GELU) v = gelu_tanh(v);
        C[(size_t)(mbase + r) * N + n] = v;
      }
    }
}

extern "C" void kernel_launch(void* const* d_in, const int* in_sizes, int n_in,
                              void* d_out, int out_size, void* d_ws, size_t ws_size,
                              hipStream_t stream)
{
  const float* x  = (const float*)d_in[0];  // [E,T,H]
  const float* w1 = (const float*)d_in[1];  // [E,H,F]
  const float* w2 = (const float*)d_in[2];  // [E,F,H]
  float*       out  = (float*)d_out;        // [E,T,H]
  float*       hbuf = (float*)d_ws;         // one [T,F] fp32 slab (32 MB), reused

  const int E = 8, T = 2048, H = 1024, F = 4096;

  for (int e = 0; e < E; ++e) {
    const float* xe  = x  + (size_t)e * T * H;
    const float* w1e = w1 + (size_t)e * H * F;
    const float* w2e = w2 + (size_t)e * F * H;
    float*       oe  = out + (size_t)e * T * H;

    // GEMM1 + GELU: [T,H]x[H,F] -> hbuf [T,F].  128x128 tiles -> 512 blocks.
    dim3 g1(F / 128, T / 128);
    hipLaunchKernelGGL((mlp_gemm_bf16x3<128, 128, 4, 2, 2, 4, true>),
                       g1, dim3(256), 0, stream, xe, w1e, hbuf, T, F, H);

    // GEMM2: [T,F]x[F,H] -> out [T,H].  64x128 tiles -> 256 blocks (fill).
    dim3 g2(H / 128, T / 64);
    hipLaunchKernelGGL((mlp_gemm_bf16x3<64, 128, 2, 4, 2, 2, false>),
                       g2, dim3(256), 0, stream, hbuf, w2e, oe, T, H, F);
  }
}
